// UnPIESTGCN_3410204033567
// MI455X (gfx1250) — compile-verified, matrix operand order checked by reading, not run on hardware
//
#include <hip/hip_runtime.h>

// ---------------------------------------------------------------------------
// UnPIESTGCN on MI455X (gfx1250): bf16 WMMA pipeline, wave32, LDS staging,
// 2x2 register-blocked WMMA tiles for operand reuse.
// Sizes: N=32 T=32 V=128 C_IN=264 C_OUT=64 K=9, TV=T*V=4096, KP=288 (pad).
// ---------------------------------------------------------------------------

typedef __attribute__((ext_vector_type(16))) __bf16 v16bf;
typedef __attribute__((ext_vector_type(8)))  __bf16 v8bf;
typedef __attribute__((ext_vector_type(8)))  float  v8f;

__device__ __forceinline__ v16bf make_frag(v8bf lo, v8bf hi) {
  return __builtin_shufflevector(lo, hi, 0,1,2,3,4,5,6,7,8,9,10,11,12,13,14,15);
}

__device__ __forceinline__ v16bf load_frag(const __bf16* row, int k0) {
  return make_frag(*(const v8bf*)(row + k0), *(const v8bf*)(row + k0 + 16));
}

__device__ __forceinline__ v8f wmma_bf16(v16bf a, v16bf b, v8f c) {
  return __builtin_amdgcn_wmma_f32_16x16x32_bf16(false, a, false, b, (short)0, c, false, false);
}

// ---------------------------------------------------------------------------
// K0: weight prep (all bf16, transposed for A-fragment contiguity).
// ---------------------------------------------------------------------------
__global__ void k_prep(const float* __restrict__ w0, const float* __restrict__ w1,
                       const float* __restrict__ wt0,
                       __bf16* __restrict__ w0t, __bf16* __restrict__ w1t,
                       __bf16* __restrict__ wtc0) {
  int i = blockIdx.x * blockDim.x + threadIdx.x;
  const int tot0 = 64 * 288, tot1 = 64 * 64, tot2 = 9 * 64 * 64;
  if (i < tot0) {
    int o = i / 288, c = i % 288;
    w0t[i] = (__bf16)((c < 264) ? w0[c * 64 + o] : 0.0f);
  } else if (i < tot0 + tot1) {
    int j = i - tot0; int o = j >> 6, c = j & 63;
    w1t[j] = (__bf16)w1[c * 64 + o];
  } else if (i < tot0 + tot1 + tot2) {
    int j = i - tot0 - tot1;
    int dt = j >> 12; int rr = j & 4095; int oc = rr >> 6; int ic = rr & 63;
    wtc0[j] = (__bf16)wt0[(dt * 64 + ic) * 64 + oc];
  }
}

// ---------------------------------------------------------------------------
// K1: concat(x,b,c) + data-BN + the reference's reshape, emitted as
//   X2T[n][tv][c2]  (bf16, row stride 288 elements = 576B, 16B aligned).
//   flat idx within n: v*8448 + t*264 + c  ==  c2*4096 + tv.
// ---------------------------------------------------------------------------
__global__ void k_bn_scatter(const float* __restrict__ x, const float* __restrict__ bx,
                             const float* __restrict__ cx,
                             const float* __restrict__ g, const float* __restrict__ bb,
                             const float* __restrict__ mm, const float* __restrict__ vv,
                             __bf16* __restrict__ x2t) {
  const int total = 32 * 128 * 32 * 264;
  for (int i = blockIdx.x * blockDim.x + threadIdx.x; i < total;
       i += gridDim.x * blockDim.x) {
    int c = i % 264;
    int r = i / 264;            // ((n*V + v)*T + t)
    int t = r & 31; r >>= 5;
    int v = r & 127;
    int n = r >> 7;
    int base = (n * 32 + t) * 128 + v;
    float val;
    if (c < 256)       val = x[(size_t)base * 256 + c];
    else if (c < 260)  val = bx[(size_t)base * 4 + (c - 256)];
    else               val = cx[(size_t)base * 4 + (c - 260)];
    int ch = v * 264 + c;
    float s = g[ch] * rsqrtf(vv[ch] + 1e-3f);
    val = (val - mm[ch]) * s + bb[ch];
    int idx = (v * 32 + t) * 264 + c;   // 0..1081343 within n
    int c2 = idx >> 12;
    int tv = idx & 4095;
    x2t[((size_t)n * 4096 + tv) * 288 + c2] = (__bf16)val;
  }
}

// ---------------------------------------------------------------------------
// K2: layer-0 projection GEMM per n: H0[64][4096] = W0T(64x264) x X2(264x4096).
//   2x2 tiles per wave (4 WMMA per k-step from 2 A + 2 B frags), W0T in LDS,
//   K tail masked by selects, streaming-B prefetch (global_prefetch_b8).
// ---------------------------------------------------------------------------
__global__ void k_proj0(const __bf16* __restrict__ x2t, const __bf16* __restrict__ w0t,
                        const float* __restrict__ bw0, __bf16* __restrict__ h0) {
  __shared__ __align__(16) __bf16 sW[64 * 288];
  {
    const int4* src = (const int4*)w0t;
    int4* dst = (int4*)sW;
    for (int i = threadIdx.x; i < (64 * 288 * 2) / 16; i += blockDim.x) dst[i] = src[i];
  }
  __syncthreads();
  int n   = blockIdx.x >> 5;          // 32 blocks per n
  int blk = blockIdx.x & 31;
  int wv = threadIdx.x >> 5, lane = threadIdx.x & 31;
  int half = lane >> 4, lm = lane & 15;
  int mtb = (wv & 1) * 2;             // m-tile base {0,2}
  int ntb = blk * 8 + (wv >> 1) * 2;  // n-tile base, 2 consecutive tiles
  const __bf16* arow0 = sW + (mtb * 16 + 16 * 0 + lm) * 288;
  const __bf16* arow1 = sW + (mtb * 16 + 16 * 1 + lm) * 288;
  const __bf16* brow0 = x2t + ((size_t)n * 4096 + ntb * 16 + lm) * 288;
  const __bf16* brow1 = brow0 + (size_t)16 * 288;
  v8f acc00 = {}, acc01 = {}, acc10 = {}, acc11 = {};
#pragma unroll
  for (int ks = 0; ks < 9; ++ks) {
    int k0 = ks * 32 + half * 8;
    __builtin_prefetch(brow0 + k0 + 64, 0, 0);
    __builtin_prefetch(brow1 + k0 + 64, 0, 0);
    v16bf a0 = load_frag(arow0, k0);
    v16bf a1 = load_frag(arow1, k0);
    v8bf b0lo = *(const v8bf*)(brow0 + k0), b0hi = *(const v8bf*)(brow0 + k0 + 16);
    v8bf b1lo = *(const v8bf*)(brow1 + k0), b1hi = *(const v8bf*)(brow1 + k0 + 16);
    if (ks == 8) {                 // K tail: valid k in [256,264)
      v8bf z = {};
      b0lo = half ? z : b0lo;      // lanes>=16 read k>=264 -> zero
      b1lo = half ? z : b1lo;
      b0hi = z; b1hi = z;          // k>=272 -> zero (A side already 0-padded)
    }
    v16bf b0 = make_frag(b0lo, b0hi);
    v16bf b1 = make_frag(b1lo, b1hi);
    acc00 = wmma_bf16(a0, b0, acc00);
    acc01 = wmma_bf16(a0, b1, acc01);
    acc10 = wmma_bf16(a1, b0, acc10);
    acc11 = wmma_bf16(a1, b1, acc11);
  }
#pragma unroll
  for (int r = 0; r < 8; ++r) {
    int o0 = (mtb + 0) * 16 + half * 8 + r;
    int o1 = (mtb + 1) * 16 + half * 8 + r;
    int c0 = (ntb + 0) * 16 + lm;
    int c1 = (ntb + 1) * 16 + lm;
    h0[((size_t)n * 64 + o0) * 4096 + c0] = (__bf16)(acc00[r] + bw0[o0]);
    h0[((size_t)n * 64 + o0) * 4096 + c1] = (__bf16)(acc01[r] + bw0[o0]);
    h0[((size_t)n * 64 + o1) * 4096 + c0] = (__bf16)(acc10[r] + bw0[o1]);
    h0[((size_t)n * 64 + o1) * 4096 + c1] = (__bf16)(acc11[r] + bw0[o1]);
  }
}

// ---------------------------------------------------------------------------
// K3: per-(n,t) graph aggregation G = H0(64x128) x (a*imp0)(128x128),
//   (a*imp0) transposed to bf16 in LDS; 2x2 tiles per wave;
//   fused BN + PReLU epilogue.
// ---------------------------------------------------------------------------
__global__ void k_agg0(const __bf16* __restrict__ h0, const float* __restrict__ a,
                       const float* __restrict__ imp,
                       const float* __restrict__ g, const float* __restrict__ bb,
                       const float* __restrict__ mm, const float* __restrict__ vv,
                       const float* __restrict__ al, __bf16* __restrict__ g0) {
  __shared__ __align__(16) __bf16 sB[128 * 136];
  int n = blockIdx.x >> 5, t = blockIdx.x & 31;
  const float* abase = a + ((size_t)n * 32 + t) * 128 * 128;
  for (int i = threadIdx.x; i < 128 * 128; i += blockDim.x) {
    int v = i >> 7, w = i & 127;
    sB[w * 136 + v] = (__bf16)(abase[i] * imp[i]);   // imp[v*128+w] == imp[i]
  }
  __syncthreads();
  int wv = threadIdx.x >> 5, lane = threadIdx.x & 31;
  int half = lane >> 4, lm = lane & 15;
  int mtb = (wv & 1) * 2;            // {0,2}
  int ntb = (wv >> 1) * 2;           // {0,2,4,6}
  const __bf16* arow0 = h0 + ((size_t)n * 64 + mtb * 16 + lm) * 4096 + t * 128;
  const __bf16* arow1 = arow0 + (size_t)16 * 4096;
  const __bf16* brow0 = sB + (ntb * 16 + lm) * 136;
  const __bf16* brow1 = brow0 + 16 * 136;
  v8f acc00 = {}, acc01 = {}, acc10 = {}, acc11 = {};
#pragma unroll
  for (int ks = 0; ks < 4; ++ks) {
    int k0 = ks * 32 + half * 8;
    v16bf a0 = load_frag(arow0, k0);
    v16bf a1 = load_frag(arow1, k0);
    v16bf b0 = load_frag(brow0, k0);
    v16bf b1 = load_frag(brow1, k0);
    acc00 = wmma_bf16(a0, b0, acc00);
    acc01 = wmma_bf16(a0, b1, acc01);
    acc10 = wmma_bf16(a1, b0, acc10);
    acc11 = wmma_bf16(a1, b1, acc11);
  }
#pragma unroll
  for (int r = 0; r < 8; ++r) {
#pragma unroll
    for (int j = 0; j < 4; ++j) {
      int o = (mtb + (j >> 1)) * 16 + half * 8 + r;
      int w = (ntb + (j & 1)) * 16 + lm;
      float acc = (j == 0) ? acc00[r] : (j == 1) ? acc01[r] : (j == 2) ? acc10[r] : acc11[r];
      float s = g[o] * rsqrtf(vv[o] + 1e-3f);
      float y = (acc - mm[o]) * s + bb[o];
      float alv = al[(o * 32 + t) * 128 + w];
      y = y >= 0.f ? y : alv * y;
      g0[((size_t)n * 64 + o) * 4096 + t * 128 + w] = (__bf16)y;
    }
  }
}

// ---------------------------------------------------------------------------
// K4: layer-0 temporal conv (K=9, SAME) per (n,v) as 9x2 accumulated WMMAs.
//   LDS tile sH[t+4][ic] zero-padded; epilogue BN; output transposed:
//   R0T[n][t*128+v][oc]  (so layer-1 B-fragments are contiguous).
// ---------------------------------------------------------------------------
__global__ void k_conv0(const __bf16* __restrict__ g0, const __bf16* __restrict__ wtc,
                        const float* __restrict__ bt,
                        const float* __restrict__ g, const float* __restrict__ bb,
                        const float* __restrict__ mm, const float* __restrict__ vv,
                        __bf16* __restrict__ r0t) {
  __shared__ __align__(16) __bf16 sH[40 * 64];
  int n = blockIdx.x >> 7, v = blockIdx.x & 127;
  for (int i = threadIdx.x; i < 40 * 64; i += blockDim.x) {
    int row = i >> 6, ic = i & 63;
    int tr = row - 4;
    __bf16 val = (__bf16)0.f;
    if (tr >= 0 && tr < 32) val = g0[((size_t)n * 64 + ic) * 4096 + tr * 128 + v];
    sH[i] = val;
  }
  __syncthreads();
  int wv = threadIdx.x >> 5, lane = threadIdx.x & 31;
  int half = lane >> 4, lm = lane & 15;
  int mt = wv & 3, nt = wv >> 2;
  int tcol = nt * 16 + lm;
  v8f acc = {};
#pragma unroll
  for (int dt = 0; dt < 9; ++dt) {
    const __bf16* arow = wtc + (dt * 64 + mt * 16 + lm) * 64;
    const __bf16* brow = sH + (tcol + dt) * 64;
#pragma unroll
    for (int ks = 0; ks < 2; ++ks) {
      int k0 = ks * 32 + half * 8;
      acc = wmma_bf16(load_frag(arow, k0), load_frag(brow, k0), acc);
    }
  }
#pragma unroll
  for (int r = 0; r < 8; ++r) {
    int oc = mt * 16 + half * 8 + r;
    float s = g[oc] * rsqrtf(vv[oc] + 1e-3f);
    float y = (acc[r] + bt[oc] - mm[oc]) * s + bb[oc];
    r0t[((size_t)n * 4096 + tcol * 128 + v) * 64 + oc] = (__bf16)y;
  }
}

// ---------------------------------------------------------------------------
// K5: layer-1 projection GEMM per n: H1[64][4096] = W1T(64x64) x R0(64x4096).
//   2x2 tiles per wave; W1T is tiny/L2-hot so A-frags come from global.
// ---------------------------------------------------------------------------
__global__ void k_proj1(const __bf16* __restrict__ r0t, const __bf16* __restrict__ w1t,
                        const float* __restrict__ bw1, __bf16* __restrict__ h1) {
  int n = blockIdx.x >> 5, blk = blockIdx.x & 31;
  int wv = threadIdx.x >> 5, lane = threadIdx.x & 31;
  int half = lane >> 4, lm = lane & 15;
  int mtb = (wv & 1) * 2;
  int ntb = blk * 8 + (wv >> 1) * 2;
  const __bf16* arow0 = w1t + (mtb * 16 + lm) * 64;
  const __bf16* arow1 = arow0 + 16 * 64;
  const __bf16* brow0 = r0t + ((size_t)n * 4096 + ntb * 16 + lm) * 64;
  const __bf16* brow1 = brow0 + (size_t)16 * 64;
  v8f acc00 = {}, acc01 = {}, acc10 = {}, acc11 = {};
#pragma unroll
  for (int ks = 0; ks < 2; ++ks) {
    int k0 = ks * 32 + half * 8;
    v16bf a0 = load_frag(arow0, k0);
    v16bf a1 = load_frag(arow1, k0);
    v16bf b0 = load_frag(brow0, k0);
    v16bf b1 = load_frag(brow1, k0);
    acc00 = wmma_bf16(a0, b0, acc00);
    acc01 = wmma_bf16(a0, b1, acc01);
    acc10 = wmma_bf16(a1, b0, acc10);
    acc11 = wmma_bf16(a1, b1, acc11);
  }
#pragma unroll
  for (int r = 0; r < 8; ++r) {
    int o0 = (mtb + 0) * 16 + half * 8 + r;
    int o1 = (mtb + 1) * 16 + half * 8 + r;
    int c0 = (ntb + 0) * 16 + lm;
    int c1 = (ntb + 1) * 16 + lm;
    h1[((size_t)n * 64 + o0) * 4096 + c0] = (__bf16)(acc00[r] + bw1[o0]);
    h1[((size_t)n * 64 + o0) * 4096 + c1] = (__bf16)(acc01[r] + bw1[o0]);
    h1[((size_t)n * 64 + o1) * 4096 + c0] = (__bf16)(acc10[r] + bw1[o1]);
    h1[((size_t)n * 64 + o1) * 4096 + c1] = (__bf16)(acc11[r] + bw1[o1]);
  }
}

// ---------------------------------------------------------------------------
// K6: pruned tail (only w=0 survives x[:,:,:,0]):
//   g1 = sum_v H1 * (a[...,0]*imp1[:,0]); BN+PReLU; temporal conv at v=0;
//   BN; + residual (layer-0 output at v=0); final (n,t,c) transpose.
// ---------------------------------------------------------------------------
__global__ void k_final(const __bf16* __restrict__ h1, const float* __restrict__ a,
                        const float* __restrict__ imp1,
                        const float* __restrict__ g1a, const float* __restrict__ b1a,
                        const float* __restrict__ m1a, const float* __restrict__ v1a,
                        const float* __restrict__ al1,
                        const float* __restrict__ wt1, const float* __restrict__ bt1,
                        const float* __restrict__ g1b, const float* __restrict__ b1b,
                        const float* __restrict__ m1b, const float* __restrict__ v1b,
                        const __bf16* __restrict__ r0t, float* __restrict__ out) {
  __shared__ float sA[32 * 128];
  __shared__ float sG[40 * 64];
  int n = blockIdx.x;
  for (int i = threadIdx.x; i < 32 * 128; i += blockDim.x) {
    int t = i >> 7, v = i & 127;
    sA[i] = a[(((size_t)n * 32 + t) * 128 + v) * 128] * imp1[v * 128];  // w = 0
  }
  for (int i = threadIdx.x; i < 40 * 64; i += blockDim.x) sG[i] = 0.f;
  __syncthreads();
  for (int i = threadIdx.x; i < 64 * 32; i += blockDim.x) {
    int c = i >> 5, t = i & 31;
    const __bf16* hrow = h1 + ((size_t)n * 64 + c) * 4096 + t * 128;
    float s = 0.f;
    for (int v = 0; v < 128; ++v) s += (float)hrow[v] * sA[t * 128 + v];
    float sc = g1a[c] * rsqrtf(v1a[c] + 1e-3f);
    float y = (s - m1a[c]) * sc + b1a[c];
    float alv = al1[(c * 32 + t) * 128];
    y = y >= 0.f ? y : alv * y;
    sG[(t + 4) * 64 + c] = y;
  }
  __syncthreads();
  for (int i = threadIdx.x; i < 64 * 32; i += blockDim.x) {
    int oc = i & 63, t = i >> 6;
    float acc = bt1[oc];
    for (int dt = 0; dt < 9; ++dt)
      for (int ic = 0; ic < 64; ++ic)
        acc += wt1[(dt * 64 + ic) * 64 + oc] * sG[(t + dt) * 64 + ic];
    float sc = g1b[oc] * rsqrtf(v1b[oc] + 1e-3f);
    float y = (acc - m1b[oc]) * sc + b1b[oc];
    y += (float)r0t[((size_t)n * 4096 + t * 128) * 64 + oc];   // residual, v=0
    out[((size_t)n * 32 + t) * 64 + oc] = y;
  }
}

// ---------------------------------------------------------------------------
extern "C" void kernel_launch(void* const* d_in, const int* in_sizes, int n_in,
                              void* d_out, int out_size, void* d_ws, size_t ws_size,
                              hipStream_t stream) {
  (void)in_sizes; (void)n_in; (void)out_size; (void)ws_size;
  const float* x    = (const float*)d_in[0];
  const float* bx   = (const float*)d_in[1];
  const float* cx   = (const float*)d_in[2];
  const float* a    = (const float*)d_in[3];
  const float* dbn_g = (const float*)d_in[4];
  const float* dbn_b = (const float*)d_in[5];
  const float* dbn_m = (const float*)d_in[6];
  const float* dbn_v = (const float*)d_in[7];
  const float* w0  = (const float*)d_in[8];
  const float* bw0 = (const float*)d_in[9];
  const float* imp0 = (const float*)d_in[10];
  const float* g0a = (const float*)d_in[11];
  const float* b0a = (const float*)d_in[12];
  const float* m0a = (const float*)d_in[13];
  const float* v0a = (const float*)d_in[14];
  const float* al0 = (const float*)d_in[15];
  const float* wt0 = (const float*)d_in[16];
  const float* bt0 = (const float*)d_in[17];
  const float* g0b = (const float*)d_in[18];
  const float* b0b = (const float*)d_in[19];
  const float* m0b = (const float*)d_in[20];
  const float* v0b = (const float*)d_in[21];
  const float* w1  = (const float*)d_in[22];
  const float* bw1 = (const float*)d_in[23];
  const float* imp1 = (const float*)d_in[24];
  const float* g1a = (const float*)d_in[25];
  const float* b1a = (const float*)d_in[26];
  const float* m1a = (const float*)d_in[27];
  const float* v1a = (const float*)d_in[28];
  const float* al1 = (const float*)d_in[29];
  const float* wt1 = (const float*)d_in[30];
  const float* bt1 = (const float*)d_in[31];
  const float* g1b = (const float*)d_in[32];
  const float* b1b = (const float*)d_in[33];
  const float* m1b = (const float*)d_in[34];
  const float* v1b = (const float*)d_in[35];
  float* out = (float*)d_out;

  char* wsp = (char*)d_ws;
  size_t off = 0;
  auto carve = [&](size_t bytes) {
    char* p = wsp + off;
    off += (bytes + 255) & ~(size_t)255;
    return p;
  };
  __bf16* x2t  = (__bf16*)carve((size_t)32 * 4096 * 288 * 2);  // 75.5 MB
  __bf16* w0t  = (__bf16*)carve((size_t)64 * 288 * 2);
  __bf16* w1t  = (__bf16*)carve((size_t)64 * 64 * 2);
  __bf16* wtc0 = (__bf16*)carve((size_t)9 * 64 * 64 * 2);
  __bf16* h0   = (__bf16*)carve((size_t)32 * 64 * 4096 * 2);   // 16.8 MB
  __bf16* g0   = (__bf16*)carve((size_t)32 * 64 * 4096 * 2);
  __bf16* r0t  = (__bf16*)carve((size_t)32 * 4096 * 64 * 2);
  __bf16* h1   = (__bf16*)carve((size_t)32 * 64 * 4096 * 2);

  k_prep<<<233, 256, 0, stream>>>(w0, w1, wt0, w0t, w1t, wtc0);
  k_bn_scatter<<<16384, 256, 0, stream>>>(x, bx, cx, dbn_g, dbn_b, dbn_m, dbn_v, x2t);
  k_proj0<<<1024, 256, 0, stream>>>(x2t, w0t, bw0, h0);
  k_agg0<<<1024, 256, 0, stream>>>(h0, a, imp0, g0a, b0a, m0a, v0a, al0, g0);
  k_conv0<<<4096, 256, 0, stream>>>(g0, wtc0, bt0, g0b, b0b, m0b, v0b, r0t);
  k_proj1<<<1024, 256, 0, stream>>>(r0t, w1t, bw1, h1);
  k_final<<<32, 256, 0, stream>>>(h1, a, imp1, g1a, b1a, m1a, v1a, al1,
                                  wt1, bt1, g1b, b1b, m1b, v1b, r0t, out);
}
